// MultiHeadAttention_53429393162348
// MI455X (gfx1250) — matrix operand order repaired and checked by reading, not compiled, not run on hardware
//
#include <hip/hip_runtime.h>

#define EMBED 1024
#define HEADS 16
#define HDIM  64
#define BATCH 2
#define SEQ   2048
#define TOKENS (BATCH * SEQ) // 4096

typedef __attribute__((ext_vector_type(16))) __bf16 v16bf;
typedef __attribute__((ext_vector_type(8)))  __bf16 v8bf;
typedef __attribute__((ext_vector_type(8)))  float  v8f;
typedef __attribute__((ext_vector_type(4)))  int    v4i;

// ---------------- async global->LDS staging (gfx1250 ASYNCcnt path) ----------

#if __has_builtin(__builtin_amdgcn_global_load_async_to_lds_b128)
#define ASYNC_LDS 1
#else
#define ASYNC_LDS 0
#endif

static __device__ __forceinline__ void cp16(__bf16* dst_lds, const __bf16* src_g) {
#if ASYNC_LDS
  __builtin_amdgcn_global_load_async_to_lds_b128(
      (__attribute__((address_space(1))) v4i*)src_g,
      (__attribute__((address_space(3))) v4i*)dst_lds, 0, 0);
#else
  *(v8bf*)dst_lds = *(const v8bf*)src_g;
#endif
}

static __device__ __forceinline__ void wait_async() {
#if ASYNC_LDS
#if __has_builtin(__builtin_amdgcn_s_wait_asynccnt)
  __builtin_amdgcn_s_wait_asynccnt(0);
#else
  asm volatile("s_wait_asynccnt 0x0" ::: "memory");
#endif
#endif
}

// ---- fragment loaders (layouts per CDNA5 ISA 7.12.2, wave32) ----------------

// 16x32 bf16 A fragment from a row-major [M x lda] matrix (global or LDS).
// element e of lane: m = row0 + (lane&15), k = k0 + (e/8)*16 + (lane>>4)*8 + (e%8)
static __device__ __forceinline__ v16bf load_a_bf16(const __bf16* base,
                                                    int lda, int row0, int k0, int lane) {
  const int m  = row0 + (lane & 15);
  const int kh = k0 + ((lane >> 4) << 3);
  const __bf16* p = base + (size_t)m * lda + kh;
  v8bf lo = *(const v8bf*)(p);        // k = kh .. kh+7
  v8bf hi = *(const v8bf*)(p + 16);   // k = kh+16 .. kh+23
  v16bf a;
#pragma unroll
  for (int i = 0; i < 8; ++i) { a[i] = lo[i]; a[i + 8] = hi[i]; }
  return a;
}

// 32x16 bf16 B fragment from a row-major [K x ldb] matrix:
// element e of lane: n = n0 + (lane&15), k = k0 + (e/8)*16 + (lane>>4)*8 + (e%8)
static __device__ __forceinline__ v16bf load_b_bf16(const __bf16* base,
                                                    int ldb, int k0, int n0, int lane) {
  const int n  = n0 + (lane & 15);
  const int kh = k0 + ((lane >> 4) << 3);
  v16bf b;
#pragma unroll
  for (int i = 0; i < 8; ++i) {
    b[i]     = base[(size_t)(kh + i) * ldb + n];
    b[i + 8] = base[(size_t)(kh + 16 + i) * ldb + n];
  }
  return b;
}

static __device__ __forceinline__ v8f wmma_bf16(v16bf a, v16bf b, v8f c) {
  return __builtin_amdgcn_wmma_f32_16x16x32_bf16(false, a, false, b, (short)0, c,
                                                 false, false);
}

// ---- converters -------------------------------------------------------------

__global__ __launch_bounds__(256) void cvt_kernel(const float* __restrict__ in,
                                                  __bf16* __restrict__ out, int n) {
  int i = blockIdx.x * blockDim.x + threadIdx.x;
  if (i < n) out[i] = (__bf16)in[i];
}

// Wt[k][n] = W[n][k]  (torch Linear: y = x @ W^T -> B matrix is W^T, K-major)
__global__ __launch_bounds__(256) void cvtT_kernel(const float* __restrict__ W,
                                                   __bf16* __restrict__ Wt) {
  int n = blockIdx.x * 16 + threadIdx.x;
  int k = blockIdx.y * 16 + threadIdx.y;
  Wt[(size_t)k * EMBED + n] = (__bf16)W[(size_t)n * EMBED + k];
}

// ---- fused QKV projection, LDS-tiled + double buffered ----------------------
// Block: 256 threads (8 waves). Block tile M=128, N=64, K-step 32.
// Wave w computes rows [w*16, w*16+16) x 64 cols of Q, K, V (12 WMMA / K-step).
// All 12 B fragments are loaded into distinct registers BEFORE the WMMAs so the
// scheduler can clause the ds_loads and use partial s_wait_dscnt instead of a
// full drain before every WMMA.
__global__ __launch_bounds__(256) void qkv_kernel(
    const __bf16* __restrict__ xbf,
    const __bf16* __restrict__ wqt, const __bf16* __restrict__ wkt,
    const __bf16* __restrict__ wvt,
    __bf16* __restrict__ Q, __bf16* __restrict__ Kt, __bf16* __restrict__ V) {
  __shared__ __bf16 As[2][128 * 32] __attribute__((aligned(16)));
  __shared__ __bf16 Bs[2][3][32 * 64] __attribute__((aligned(16)));
  const int tid  = threadIdx.x;
  const int lane = tid & 31;
  const int wave = tid >> 5;
  const int n0   = blockIdx.x * 64;
  const int gm0  = blockIdx.y * 128;

  auto stage = [&](int buf, int k0) {
    // A tile 128x32 bf16 = 512 x 16B chunks, 2 per thread (row-major, lda=32)
#pragma unroll
    for (int i = 0; i < 2; ++i) {
      const int c = tid * 2 + i;
      const int row = c >> 2, colc = (c & 3) * 8;
      cp16(&As[buf][row * 32 + colc],
           xbf + (size_t)(gm0 + row) * EMBED + k0 + colc);
    }
    // B tiles 3 x (32x64) = 256 chunks each, 1 per thread per matrix (ldb=64)
    const int row = tid >> 3, colc = (tid & 7) * 8;
    cp16(&Bs[buf][0][row * 64 + colc], wqt + (size_t)(k0 + row) * EMBED + n0 + colc);
    cp16(&Bs[buf][1][row * 64 + colc], wkt + (size_t)(k0 + row) * EMBED + n0 + colc);
    cp16(&Bs[buf][2][row * 64 + colc], wvt + (size_t)(k0 + row) * EMBED + n0 + colc);
  };

  v8f accQ[4] = {}, accK[4] = {}, accV[4] = {};

  stage(0, 0);
  wait_async();
  __syncthreads();

  const int nsteps = EMBED / 32;
  for (int step = 0; step < nsteps; ++step) {
    const int buf = step & 1;
    if (step + 1 < nsteps) stage(buf ^ 1, (step + 1) * 32);  // overlap with WMMAs

    // Load A + all 12 B fragments into distinct registers first...
    const v16bf a = load_a_bf16(As[buf], 32, wave * 16, 0, lane);
    v16bf bq[4], bk[4], bv[4];
#pragma unroll
    for (int nt = 0; nt < 4; ++nt) {
      bq[nt] = load_b_bf16(Bs[buf][0], 64, 0, nt * 16, lane);
      bk[nt] = load_b_bf16(Bs[buf][1], 64, 0, nt * 16, lane);
      bv[nt] = load_b_bf16(Bs[buf][2], 64, 0, nt * 16, lane);
    }
    // ...then issue the 12 WMMAs back-to-back.
#pragma unroll
    for (int nt = 0; nt < 4; ++nt) {
      accQ[nt] = wmma_bf16(a, bq[nt], accQ[nt]);
      accK[nt] = wmma_bf16(a, bk[nt], accK[nt]);
      accV[nt] = wmma_bf16(a, bv[nt], accV[nt]);
    }
    wait_async();     // next panel landed in LDS
    __syncthreads();  // all waves done reading buf; buf^1 visible to everyone
  }

  const int half = lane >> 4, nn = lane & 15;
#pragma unroll
  for (int nt = 0; nt < 4; ++nt) {
    const int col = n0 + nt * 16 + nn;
    const int h = col >> 6, d = col & (HDIM - 1);
#pragma unroll
    for (int r = 0; r < 8; ++r) {
      const int gm = gm0 + wave * 16 + r + 8 * half;
      const int b = gm >> 11, tok = gm & (SEQ - 1);
      const int bh = b * HEADS + h;
      Q[((size_t)bh * SEQ + tok) * HDIM + d]  = (__bf16)accQ[nt][r];
      Kt[((size_t)bh * HDIM + d) * SEQ + tok] = (__bf16)accK[nt][r];
      V[((size_t)bh * SEQ + tok) * HDIM + d]  = (__bf16)accV[nt][r];
    }
  }
}

// ---- flash attention: one wave per (b, h, 16-query tile) --------------------
__global__ __launch_bounds__(32) void attn_kernel(
    const __bf16* __restrict__ Q, const __bf16* __restrict__ Kt,
    const __bf16* __restrict__ V, __bf16* __restrict__ O) {
  __shared__ __bf16 Plds[16 * 32] __attribute__((aligned(16)));
  const int lane = threadIdx.x & 31;
  const int half = lane >> 4;
  const int nn   = lane & 15;
  const int q0 = blockIdx.x * 16;
  const int bh = blockIdx.z * HEADS + blockIdx.y;
  const __bf16* Qh = Q  + (size_t)bh * SEQ * HDIM;
  const __bf16* Kh = Kt + (size_t)bh * HDIM * SEQ;
  const __bf16* Vh = V  + (size_t)bh * SEQ * HDIM;

  const v16bf aq0 = load_a_bf16(Qh, HDIM, q0, 0, lane);
  const v16bf aq1 = load_a_bf16(Qh, HDIM, q0, 32, lane);

  v8f o0 = {}, o1 = {}, o2 = {}, o3 = {};
  float rmax[8], rsum[8];
#pragma unroll
  for (int r = 0; r < 8; ++r) { rmax[r] = -__builtin_inff(); rsum[r] = 0.f; }

  const int nChunks = (q0 + 47) >> 5;  // causal: keys 0 .. q0+15
  for (int c = 0; c < nChunks; ++c) {
    const int kk0 = c * 32;
    // Issue ALL loads for this chunk up front: K-frags feed the score WMMAs,
    // V-frags are consumed only after softmax (loads overlap the VALU work).
    v16bf b0  = load_b_bf16(Kh, SEQ, 0,  kk0,      lane);
    v16bf b1  = load_b_bf16(Kh, SEQ, 32, kk0,      lane);
    v16bf b2  = load_b_bf16(Kh, SEQ, 0,  kk0 + 16, lane);
    v16bf b3  = load_b_bf16(Kh, SEQ, 32, kk0 + 16, lane);
    v16bf bv0 = load_b_bf16(Vh, HDIM, kk0, 0,  lane);
    v16bf bv1 = load_b_bf16(Vh, HDIM, kk0, 16, lane);
    v16bf bv2 = load_b_bf16(Vh, HDIM, kk0, 32, lane);
    v16bf bv3 = load_b_bf16(Vh, HDIM, kk0, 48, lane);

    v8f s0 = {}, s1 = {};
    s0 = wmma_bf16(aq0, b0, s0);
    s0 = wmma_bf16(aq1, b1, s0);
    s1 = wmma_bf16(aq0, b2, s1);
    s1 = wmma_bf16(aq1, b3, s1);

    const bool lastChunk = (c == nChunks - 1);  // only chunk crossing diagonal
    __syncthreads();                            // Plds WAR vs previous iteration
    float scale[8];
#pragma unroll
    for (int r = 0; r < 8; ++r) {
      const int gq = q0 + r + 8 * half;         // C layout: row = r + 8*(lane/16)
      float v0 = s0[r] * 0.125f;                // 1/sqrt(64)
      float v1 = s1[r] * 0.125f;
      if (lastChunk) {
        if (kk0 + nn > gq)      v0 = -__builtin_inff();
        if (kk0 + 16 + nn > gq) v1 = -__builtin_inff();
      }
      float mx = fmaxf(v0, v1);
#pragma unroll
      for (int off = 1; off < 16; off <<= 1) mx = fmaxf(mx, __shfl_xor(mx, off, 16));
      const float nmax = fmaxf(rmax[r], mx);
      const float sc = __expf(rmax[r] - nmax);  // exp(-inf)=0 on first chunk
      rmax[r] = nmax;
      const float p0 = __expf(v0 - nmax);
      const float p1 = __expf(v1 - nmax);
      float ps = p0 + p1;
#pragma unroll
      for (int off = 1; off < 16; off <<= 1) ps += __shfl_xor(ps, off, 16);
      rsum[r] = rsum[r] * sc + ps;
      scale[r] = sc;
      // stage P (16 queries x 32 keys) row-major for re-layout into A-frag
      Plds[(r + 8 * half) * 32 + nn]      = (__bf16)p0;
      Plds[(r + 8 * half) * 32 + 16 + nn] = (__bf16)p1;
    }
#pragma unroll
    for (int r = 0; r < 8; ++r) {
      o0[r] *= scale[r]; o1[r] *= scale[r];
      o2[r] *= scale[r]; o3[r] *= scale[r];
    }
    __syncthreads();                            // Plds stores visible to all lanes
    v16bf ap = load_a_bf16(Plds, 32, 0, 0, lane);
    o0 = wmma_bf16(ap, bv0, o0);
    o1 = wmma_bf16(ap, bv1, o1);
    o2 = wmma_bf16(ap, bv2, o2);
    o3 = wmma_bf16(ap, bv3, o3);
  }
  // normalize rows, store attn output token-major [4096 x 1024] bf16
#pragma unroll
  for (int r = 0; r < 8; ++r) {
    const float inv = 1.0f / rsum[r];
    const int gq = q0 + r + 8 * half;
    __bf16* row = O + ((size_t)blockIdx.z * SEQ + gq) * EMBED + blockIdx.y * HDIM;
    row[0 + nn]  = (__bf16)(o0[r] * inv);
    row[16 + nn] = (__bf16)(o1[r] * inv);
    row[32 + nn] = (__bf16)(o2[r] * inv);
    row[48 + nn] = (__bf16)(o3[r] * inv);
  }
}

// ---- output projection, LDS-tiled + double buffered: out = attn @ Wo^T + bo -
__global__ __launch_bounds__(256) void proj_kernel(
    const __bf16* __restrict__ attn, const __bf16* __restrict__ wot,
    const float* __restrict__ bo, float* __restrict__ out) {
  __shared__ __bf16 As[2][128 * 32] __attribute__((aligned(16)));
  __shared__ __bf16 Bs[2][32 * 64] __attribute__((aligned(16)));
  const int tid  = threadIdx.x;
  const int lane = tid & 31;
  const int wave = tid >> 5;
  const int n0   = blockIdx.x * 64;
  const int gm0  = blockIdx.y * 128;

  auto stage = [&](int buf, int k0) {
#pragma unroll
    for (int i = 0; i < 2; ++i) {
      const int c = tid * 2 + i;
      const int row = c >> 2, colc = (c & 3) * 8;
      cp16(&As[buf][row * 32 + colc],
           attn + (size_t)(gm0 + row) * EMBED + k0 + colc);
    }
    const int row = tid >> 3, colc = (tid & 7) * 8;
    cp16(&Bs[buf][row * 64 + colc], wot + (size_t)(k0 + row) * EMBED + n0 + colc);
  };

  v8f acc[4] = {};

  stage(0, 0);
  wait_async();
  __syncthreads();

  const int nsteps = EMBED / 32;
  for (int step = 0; step < nsteps; ++step) {
    const int buf = step & 1;
    if (step + 1 < nsteps) stage(buf ^ 1, (step + 1) * 32);
    const v16bf a = load_a_bf16(As[buf], 32, wave * 16, 0, lane);
    v16bf bw[4];
#pragma unroll
    for (int nt = 0; nt < 4; ++nt) bw[nt] = load_b_bf16(Bs[buf], 64, 0, nt * 16, lane);
#pragma unroll
    for (int nt = 0; nt < 4; ++nt) acc[nt] = wmma_bf16(a, bw[nt], acc[nt]);
    wait_async();
    __syncthreads();
  }

  const int half = lane >> 4, nn = lane & 15;
#pragma unroll
  for (int nt = 0; nt < 4; ++nt) {
    const int col = n0 + nt * 16 + nn;
    const float bias = bo[col];
#pragma unroll
    for (int r = 0; r < 8; ++r) {
      out[(size_t)(gm0 + wave * 16 + r + 8 * half) * EMBED + col] = acc[nt][r] + bias;
    }
  }
}

// ---- host launch ------------------------------------------------------------

extern "C" void kernel_launch(void* const* d_in, const int* in_sizes, int n_in,
                              void* d_out, int out_size, void* d_ws, size_t ws_size,
                              hipStream_t stream) {
  (void)in_sizes; (void)n_in; (void)out_size; (void)ws_size;
  const float* x  = (const float*)d_in[0];
  // d_in[1] = causal mask (int) -- causality hardcoded in attn_kernel
  const float* Wq = (const float*)d_in[2];
  const float* Wk = (const float*)d_in[3];
  const float* Wv = (const float*)d_in[4];
  const float* Wo = (const float*)d_in[5];
  const float* bo = (const float*)d_in[6];
  float* out = (float*)d_out;

  char* ws = (char*)d_ws;
  size_t off = 0;
  auto alloc = [&](size_t bytes) {
    char* p = ws + off;
    off += (bytes + 255) & ~(size_t)255;
    return p;
  };
  __bf16* xbf  = (__bf16*)alloc((size_t)TOKENS * EMBED * 2);
  __bf16* wqt  = (__bf16*)alloc((size_t)EMBED * EMBED * 2);
  __bf16* wkt  = (__bf16*)alloc((size_t)EMBED * EMBED * 2);
  __bf16* wvt  = (__bf16*)alloc((size_t)EMBED * EMBED * 2);
  __bf16* wot  = (__bf16*)alloc((size_t)EMBED * EMBED * 2);
  __bf16* Qb   = (__bf16*)alloc((size_t)TOKENS * EMBED * 2);   // [bh][N][64]
  __bf16* Ktb  = (__bf16*)alloc((size_t)TOKENS * EMBED * 2);   // [bh][64][N]
  __bf16* Vb   = (__bf16*)alloc((size_t)TOKENS * EMBED * 2);
  __bf16* attn = (__bf16*)alloc((size_t)TOKENS * EMBED * 2);   // [4096 x 1024]

  const int nx = TOKENS * EMBED;
  cvt_kernel<<<nx / 256, 256, 0, stream>>>(x, xbf, nx);
  dim3 tB(16, 16), tG(EMBED / 16, EMBED / 16);
  cvtT_kernel<<<tG, tB, 0, stream>>>(Wq, wqt);
  cvtT_kernel<<<tG, tB, 0, stream>>>(Wk, wkt);
  cvtT_kernel<<<tG, tB, 0, stream>>>(Wv, wvt);
  cvtT_kernel<<<tG, tB, 0, stream>>>(Wo, wot);

  qkv_kernel<<<dim3(EMBED / 64, TOKENS / 128), 256, 0, stream>>>(
      xbf, wqt, wkt, wvt, Qb, Ktb, Vb);

  attn_kernel<<<dim3(SEQ / 16, HEADS, BATCH), 32, 0, stream>>>(Qb, Ktb, Vb, attn);

  proj_kernel<<<dim3(EMBED / 64, TOKENS / 128), 256, 0, stream>>>(attn, wot, bo, out);
}
